// BiLstm_64209761075484
// MI455X (gfx1250) — matrix operand order repaired and checked by reading, not compiled, not run on hardware
//
#include <hip/hip_runtime.h>
#include <hip/hip_bf16.h>

// Problem constants: B=64, T=256, D=512, H=512
#define BB 64
#define TT 256
#define DD 512
#define HH 512
#define GG 2048   // 4*H gate width
#define KP 1024   // D+H concat input width

typedef __bf16 v16bf __attribute__((ext_vector_type(16)));
typedef float  v8f   __attribute__((ext_vector_type(8)));
typedef unsigned int u32x4 __attribute__((ext_vector_type(4)));

union Frag {
    u32x4 u[2];
    v16bf v;
};

// ---- forced global (addrspace 1) accesses: generic==global bit pattern ----
__device__ __forceinline__ u32x4 gload16(const void* p) {
    return *(const __attribute__((address_space(1))) u32x4*)(unsigned long long)p;
}
__device__ __forceinline__ float gloadf(const float* p) {
    return *(const __attribute__((address_space(1))) float*)(unsigned long long)p;
}
__device__ __forceinline__ void gstoref(float* p, float v) {
    *(__attribute__((address_space(1))) float*)(unsigned long long)p = v;
}
__device__ __forceinline__ void gstoreu16(unsigned short* p, unsigned short v) {
    *(__attribute__((address_space(1))) unsigned short*)(unsigned long long)p = v;
}

__device__ __forceinline__ unsigned short f2bf(float f) {
    unsigned int u = __builtin_bit_cast(unsigned int, f);
    u += 0x7FFFu + ((u >> 16) & 1u);   // round-to-nearest-even
    return (unsigned short)(u >> 16);
}
__device__ __forceinline__ float sigmoidf_(float x) {
    return 1.0f / (1.0f + __expf(-x));
}
__device__ __forceinline__ float tanhf_(float x) {
    float e = __expf(2.0f * x);
    return 1.0f - 2.0f / (e + 1.0f);
}

// ---------------- prep kernels ----------------
__global__ void init_state(unsigned short* hG, unsigned* sync) {
    int i = blockIdx.x * blockDim.x + threadIdx.x;
    const int n = 2 * 2 * BB * HH;          // two parities, two dirs
    for (; i < n; i += gridDim.x * blockDim.x) hG[i] = 0;
    if (blockIdx.x == 0 && threadIdx.x == 0) { sync[0] = 0u; sync[1] = 0u; }
}

__global__ void cvt_x_bf16(const float* __restrict__ x,
                           unsigned short* __restrict__ xb, int n) {
    int i = blockIdx.x * blockDim.x + threadIdx.x;
    int stride = gridDim.x * blockDim.x;
    for (; i < n; i += stride) xb[i] = f2bf(x[i]);
}

__global__ void transpose_w(const float* __restrict__ W,
                            unsigned short* __restrict__ WT) {
    int i = blockIdx.x * blockDim.x + threadIdx.x;  // over GG*KP
    if (i >= GG * KP) return;
    int n = i >> 10;       // KP = 1024
    int k = i & 1023;
    WT[i] = f2bf(W[(size_t)k * GG + n]);
}

// ---------------- phase 1: gates_x = x @ Wx + b (both dirs, all t) ----------------
// wave macro-tile: M=32 (2 subtiles) x N=64 (4 subtiles), K=512
// wid = ((dir*256 + t)*2 + mt)*32 + nt ; 32768 waves = 4096 blocks x 256 thr
__launch_bounds__(256)
__global__ void xgates_gemm(const unsigned short* __restrict__ xb,
                            const unsigned short* __restrict__ WT,
                            const float* __restrict__ bfw,
                            const float* __restrict__ bbw,
                            float* __restrict__ gx) {
    const int lane = threadIdx.x & 31;
    const int lhi = lane >> 4, llo = lane & 15;
    const int ksel = lhi * 8, bsel = lhi * 16;

    int wid = blockIdx.x * 8 + (threadIdx.x >> 5);
    const int nt = wid & 31;  wid >>= 5;
    const int mt = wid & 1;   wid >>= 1;
    const int t  = wid & 255; wid >>= 8;
    const int dir = wid;

    const float* bia = dir ? bbw : bfw;
    const unsigned short* Wt = WT + (size_t)dir * GG * KP;

    v8f acc[2][4];
#pragma unroll
    for (int ms = 0; ms < 2; ++ms)
#pragma unroll
        for (int sub = 0; sub < 4; ++sub) {
            float b = gloadf(&bia[nt * 64 + sub * 16 + llo]);
            v8f z;
#pragma unroll
            for (int e = 0; e < 8; ++e) z[e] = b;
            acc[ms][sub] = z;
        }

    for (int kt = 0; kt < 16; ++kt) {
        const int k0 = kt * 32;
        Frag a[2];
#pragma unroll
        for (int ms = 0; ms < 2; ++ms) {
            const unsigned short* ap =
                xb + ((size_t)(mt * 32 + ms * 16 + llo) * TT + t) * DD + k0 + ksel;
            a[ms].u[0] = gload16(ap);
            a[ms].u[1] = gload16(ap + 16);
        }
#pragma unroll
        for (int sub = 0; sub < 4; ++sub) {
            const unsigned short* bp =
                Wt + (size_t)(nt * 64 + sub * 16 + llo) * KP + DD + k0 + ksel;
            Frag bm;
            bm.u[0] = gload16(bp);
            bm.u[1] = gload16(bp + 16);
#pragma unroll
            for (int ms = 0; ms < 2; ++ms)
                acc[ms][sub] = __builtin_amdgcn_wmma_f32_16x16x32_bf16(
                    false, a[ms].v, false, bm.v, (short)0, acc[ms][sub],
                    false, false);
        }
    }

    float* gbase = gx + (size_t)(dir * TT + t) * BB * GG;
#pragma unroll
    for (int ms = 0; ms < 2; ++ms)
#pragma unroll
        for (int sub = 0; sub < 4; ++sub)
#pragma unroll
            for (int v = 0; v < 8; ++v) {
                const int m = mt * 32 + ms * 16 + v + 8 * lhi;
                const int n = nt * 64 + sub * 16 + llo;
                gstoref(&gbase[(size_t)m * GG + n], acc[ms][sub][v]);
            }
}

// ---------------- phase 2: serial scan, h-part only (K=512) ----------------
// grid = 8 blocks: dir(2) x j-slice(4 of 128 cols). M=64 (full batch) per block.
// 32 waves: msub(4 of 16 rows) x jt(8 of 16 cols). 4 gate tiles per wave.
// h exchanged via double-buffered global hG + 8-block grid barrier per step.
__launch_bounds__(1024, 1)
__global__ void lstm_scan(const float* __restrict__ mask,
                          const unsigned short* __restrict__ WT,
                          const float* __restrict__ gx,
                          unsigned short* hG,          // [2][2][64][512] bf16
                          unsigned* sync,              // {count, generation}
                          float* __restrict__ out) {
    const int dir  = blockIdx.x >> 2;
    const int jblk = blockIdx.x & 3;
    const unsigned short* Wt0 = WT + (size_t)dir * GG * KP;

    __shared__ __align__(16) unsigned short h_lds[BB * HH];  // 64 KB bf16

    const int tid  = threadIdx.x;
    const int w    = tid >> 5;
    const int lane = tid & 31;
    const int lhi  = lane >> 4, llo = lane & 15;
    const int ksel = lhi * 8, bsel = lhi * 16;
    const int msub = w >> 3;                 // 0..3  -> rows msub*16..+15
    const int jt   = w & 7;                  // 0..7  -> 16-col slice
    const int jloc = jblk * 128 + jt * 16 + llo;   // h column in [0,512)

    // c,h state in registers; element (m = msub*16 + v + 8*lhi, jloc)
    float creg[8], hreg[8];
#pragma unroll
    for (int v = 0; v < 8; ++v) { creg[v] = 0.0f; hreg[v] = 0.0f; }

    for (int s = 0; s < TT; ++s) {
        const int t = dir ? (TT - 1 - s) : s;
        const int p = s & 1;

        // stage h_{s} (this dir) from global into LDS
        {
            const unsigned short* src = hG + ((size_t)p * 2 + dir) * (BB * HH);
            for (int i = tid * 8; i < BB * HH; i += 1024 * 8)
                *(u32x4*)&h_lds[i] = gload16(src + i);
        }
        __syncthreads();

        // launder weight offset: keep weight loads inside the step loop
        size_t woff = 0;
        asm volatile("" : "+s"(woff));
        const unsigned short* Wt = Wt0 + woff;

        // acc init = gates_x (contains x@Wx + bias)
        const float* gxt = gx + (size_t)(dir * TT + t) * BB * GG;
        v8f acc[4];
#pragma unroll
        for (int g = 0; g < 4; ++g) {
            const int n = g * HH + jloc;
            v8f z;
#pragma unroll
            for (int v = 0; v < 8; ++v)
                z[v] = gloadf(&gxt[(size_t)(msub * 16 + v + 8 * lhi) * GG + n]);
            acc[g] = z;
        }

        // h-part GEMM: A = h (LDS rows msub*16..+15), B = Wh^T rows g*512+jloc
        for (int kt = 0; kt < 16; ++kt) {
            const int k0 = kt * 32;
            Frag a;
            const unsigned short* ap = &h_lds[(msub * 16 + llo) * HH + k0 + ksel];
            a.u[0] = *(const u32x4*)ap;
            a.u[1] = *(const u32x4*)(ap + 16);
#pragma unroll
            for (int g = 0; g < 4; ++g) {
                const unsigned short* bp =
                    Wt + (size_t)(g * HH + jloc) * KP + k0 + ksel;
                Frag bm;
                bm.u[0] = gload16(bp);
                bm.u[1] = gload16(bp + 16);
                acc[g] = __builtin_amdgcn_wmma_f32_16x16x32_bf16(
                    false, a.v, false, bm.v, (short)0, acc[g], false, false);
            }
        }

        // elementwise recurrence + h broadcast + output
        unsigned short* hdst = hG + ((size_t)(p ^ 1) * 2 + dir) * (BB * HH);
#pragma unroll
        for (int v = 0; v < 8; ++v) {
            const int m = msub * 16 + v + 8 * lhi;
            float iv = sigmoidf_(acc[0][v]);
            float fv = sigmoidf_(acc[1][v]);
            float ov = sigmoidf_(acc[2][v]);
            ov = (ov > 0.4f) ? ov : 0.0f;
            float cg = tanhf_(acc[3][v]);
            float cn = fv * creg[v] + iv * cg;
            float hn = ov * tanhf_(cn);
            float mk = gloadf(&mask[(size_t)m * TT + t]);
            float hb2 = mk * hn + (1.0f - mk) * hreg[v];
            float cb2 = mk * cn + (1.0f - mk) * creg[v];
            creg[v] = cb2;
            hreg[v] = hb2;
            gstoreu16(&hdst[(size_t)m * HH + jloc], f2bf(hb2));
            gstoref(&out[((size_t)m * TT + t) * (2 * HH) + dir * HH + jloc], hb2);
        }

        // ---- 8-block grid barrier (arrive + generation spin) ----
        __syncthreads();
        if (tid == 0) {
            unsigned g = __hip_atomic_load(&sync[1], __ATOMIC_RELAXED,
                                           __HIP_MEMORY_SCOPE_AGENT);
            __threadfence();
            if (__hip_atomic_fetch_add(&sync[0], 1u, __ATOMIC_ACQ_REL,
                                       __HIP_MEMORY_SCOPE_AGENT) == 8u - 1u) {
                __hip_atomic_store(&sync[0], 0u, __ATOMIC_RELAXED,
                                   __HIP_MEMORY_SCOPE_AGENT);
                __hip_atomic_store(&sync[1], g + 1u, __ATOMIC_RELEASE,
                                   __HIP_MEMORY_SCOPE_AGENT);
            } else {
                while (__hip_atomic_load(&sync[1], __ATOMIC_ACQUIRE,
                                         __HIP_MEMORY_SCOPE_AGENT) == g)
                    __builtin_amdgcn_s_sleep(2);
            }
        }
        __syncthreads();
    }
}

extern "C" void kernel_launch(void* const* d_in, const int* in_sizes, int n_in,
                              void* d_out, int out_size, void* d_ws, size_t ws_size,
                              hipStream_t stream) {
    const float* x    = (const float*)d_in[0];
    const float* mask = (const float*)d_in[1];
    const float* Wf   = (const float*)d_in[2];
    const float* bf   = (const float*)d_in[3];
    const float* Wb   = (const float*)d_in[4];
    const float* bb   = (const float*)d_in[5];
    float* out = (float*)d_out;

    // workspace layout
    char* ws = (char*)d_ws;
    unsigned short* WT = (unsigned short*)ws;                       // 8 MB
    unsigned short* xb = (unsigned short*)(ws + 8388608);           // 16.8 MB
    unsigned short* hG = (unsigned short*)(ws + 25165824);          // 256 KB
    unsigned*     sync = (unsigned*)(ws + 25427968);                // 256 B
    float*          gx = (float*)(ws + 25428224);                   // 268 MB

    init_state<<<64, 256, 0, stream>>>(hG, sync);
    const int wtn = GG * KP;
    transpose_w<<<(wtn + 255) / 256, 256, 0, stream>>>(Wf, WT);
    transpose_w<<<(wtn + 255) / 256, 256, 0, stream>>>(Wb, WT + wtn);
    const int xn = BB * TT * DD;
    cvt_x_bf16<<<(xn + 255) / 256, 256, 0, stream>>>(x, xb, xn);

    xgates_gemm<<<4096, 256, 0, stream>>>(xb, WT, bf, bb, gx);
    lstm_scan<<<8, 1024, 0, stream>>>(mask, WT, gx, hG, sync, out);
}
